// GAT_33397665694065
// MI455X (gfx1250) — matrix operand order repaired and checked by reading, not compiled, not run on hardware
//
#include <hip/hip_runtime.h>

// ---------------- problem constants (match reference) ----------------
constexpr int N_NODES = 50000;
constexpr int N_EDGES = 1600000;
constexpr int EN      = N_EDGES + N_NODES;   // edges + self loops
constexpr int DH      = 128;
constexpr int DOUT    = 100;
constexpr int DOUTP   = 112;                 // DOUT padded to 16
constexpr float BN_EPS    = 1e-5f;
constexpr float NEG_SLOPE = 0.2f;

typedef __attribute__((ext_vector_type(16))) _Float16 v16h;
typedef __attribute__((ext_vector_type(8)))  _Float16 v8h;
typedef __attribute__((ext_vector_type(8)))  float    v8f;

#define DEVFN __device__ __forceinline__

DEVFN float warpRedMax(float v) {
#pragma unroll
  for (int o = 16; o > 0; o >>= 1) v = fmaxf(v, __shfl_xor(v, o, 32));
  return v;
}
DEVFN float warpRedSum(float v) {
#pragma unroll
  for (int o = 16; o > 0; o >>= 1) v += __shfl_xor(v, o, 32);
  return v;
}

// ---------------- CSR construction (graph constant across layers) ----------------
__global__ void k_zero_counts(int* __restrict__ counts) {
  int i = blockIdx.x * blockDim.x + threadIdx.x;
  if (i < N_NODES) counts[i] = 0;
}

__global__ void k_histogram(const int* __restrict__ ei, int* __restrict__ counts) {
  int j = blockIdx.x * blockDim.x + threadIdx.x;
  if (j >= EN) return;
  int dst = (j < N_EDGES) ? ei[N_EDGES + j] : (j - N_EDGES);
  atomicAdd(&counts[dst], 1);
}

__global__ void k_scan(const int* __restrict__ counts, int* __restrict__ offsets,
                       int* __restrict__ cursor) {
  __shared__ int sums[1024];
  int t = threadIdx.x;
  const int chunk = (N_NODES + 1023) / 1024;
  int beg = t * chunk;
  int end = beg + chunk; if (end > N_NODES) end = N_NODES;
  if (beg > N_NODES) beg = N_NODES;
  int s = 0;
  for (int i = beg; i < end; ++i) s += counts[i];
  sums[t] = s;
  __syncthreads();
  for (int off = 1; off < 1024; off <<= 1) {
    int v = (t >= off) ? sums[t - off] : 0;
    __syncthreads();
    sums[t] += v;
    __syncthreads();
  }
  int run = sums[t] - s;  // exclusive prefix of this thread's chunk
  for (int i = beg; i < end; ++i) {
    offsets[i] = run; cursor[i] = run; run += counts[i];
  }
  if (t == 1023) offsets[N_NODES] = sums[1023];
}

__global__ void k_scatter(const int* __restrict__ ei, int* __restrict__ cursor,
                          int* __restrict__ ssrc) {
  int j = blockIdx.x * blockDim.x + threadIdx.x;
  if (j >= EN) return;
  int src, dst;
  if (j < N_EDGES) { src = ei[j]; dst = ei[N_EDGES + j]; }
  else             { src = j - N_EDGES; dst = src; }
  int pos = atomicAdd(&cursor[dst], 1);
  ssrc[pos] = src;
}

// ---------------- weight prep: transpose + f32->f16 (+N padding) ----------------
__global__ void k_prep_weight(const float* __restrict__ W, _Float16* __restrict__ Wt,
                              int din, int dout, int dpad) {
  int i = blockIdx.x * blockDim.x + threadIdx.x;
  if (i >= dpad * din) return;
  int n = i / din, k = i - n * din;
  float v = (n < dout) ? W[k * dout + n] : 0.f;
  Wt[(size_t)n * din + k] = (_Float16)v;
}

// ---------------- layer-1 GEMM (K=3, trivial) ----------------
__global__ void k_l1_gemm(const float* __restrict__ pos, const float* __restrict__ W1,
                          float* __restrict__ h) {
  int i = blockIdx.x * blockDim.x + threadIdx.x;
  if (i >= N_NODES * DH) return;
  int m = i >> 7, n = i & 127;
  float acc = pos[m * 3 + 0] * W1[n]
            + pos[m * 3 + 1] * W1[DH + n]
            + pos[m * 3 + 2] * W1[2 * DH + n];
  h[i] = acc;
}

// ---------------- attention logits: al_s = h.a_src, al_d = h.a_dst ----------------
__global__ void k_al(const float* __restrict__ h, const float* __restrict__ asrc,
                     const float* __restrict__ adst, float* __restrict__ als,
                     float* __restrict__ ald) {
  int wid  = (blockIdx.x * blockDim.x + threadIdx.x) >> 5;
  int lane = threadIdx.x & 31;
  if (wid >= N_NODES) return;
  float4 hv = ((const float4*)(h + (size_t)wid * DH))[lane];
  float4 s4 = ((const float4*)asrc)[lane];
  float4 d4 = ((const float4*)adst)[lane];
  float ss = hv.x * s4.x + hv.y * s4.y + hv.z * s4.z + hv.w * s4.w;
  float dd = hv.x * d4.x + hv.y * d4.y + hv.z * d4.z + hv.w * d4.w;
  ss = warpRedSum(ss);
  dd = warpRedSum(dd);
  if (lane == 0) { als[wid] = ss; ald[wid] = dd; }
}

// ---------------- GAT aggregation: one wave per dst node, atomic-free ----------------
// fused epilogue: out = bn( relu( sum_j alpha_j h[src_j] + bias ) ) -> f16 for next GEMM
__global__ void k_aggregate(const int* __restrict__ offsets, const int* __restrict__ ssrc,
                            float* __restrict__ ews,
                            const float* __restrict__ als, const float* __restrict__ ald_,
                            const float* __restrict__ h, const float* __restrict__ bias,
                            const float* __restrict__ bng, const float* __restrict__ bnb,
                            _Float16* __restrict__ xout) {
  int wid  = (blockIdx.x * blockDim.x + threadIdx.x) >> 5;
  int lane = threadIdx.x & 31;
  if (wid >= N_NODES) return;
  int beg = offsets[wid], end = offsets[wid + 1];
  float ald = ald_[wid];

  // phase 1: e_j = LeakyReLU(al_s[src]+al_d[dst]); segment max
  float mx = -3.4e38f;
  for (int i = beg + lane; i < end; i += 32) {
    float z = als[ssrc[i]] + ald;
    float e = (z >= 0.f) ? z : z * NEG_SLOPE;
    ews[i] = e;
    mx = fmaxf(mx, e);
  }
  mx = warpRedMax(mx);

  // phase 2: ex = exp(e-m); segment sum
  float sum = 0.f;
  for (int i = beg + lane; i < end; i += 32) {
    float ex = __expf(ews[i] - mx);
    ews[i] = ex;
    sum += ex;
  }
  sum = warpRedSum(sum);
  float rden = 1.f / (sum + 1e-16f);

  // phase 3: acc = sum_j alpha_j * h[src_j]; lane owns 4 contiguous features
  float4 acc = {0.f, 0.f, 0.f, 0.f};
  for (int i = beg; i < end; ++i) {
    float alpha = ews[i] * rden;
    float4 hv = ((const float4*)(h + (size_t)ssrc[i] * DH))[lane];
    acc.x += alpha * hv.x; acc.y += alpha * hv.y;
    acc.z += alpha * hv.z; acc.w += alpha * hv.w;
  }

  // epilogue: +bias, relu, bn(eval), emit f16 activations
  float4 bv = ((const float4*)bias)[lane];
  float4 g4 = ((const float4*)bng)[lane];
  float4 b4 = ((const float4*)bnb)[lane];
  const float sc = rsqrtf(1.f + BN_EPS);
  float o0 = fmaxf(acc.x + bv.x, 0.f) * (g4.x * sc) + b4.x;
  float o1 = fmaxf(acc.y + bv.y, 0.f) * (g4.y * sc) + b4.y;
  float o2 = fmaxf(acc.z + bv.z, 0.f) * (g4.z * sc) + b4.z;
  float o3 = fmaxf(acc.w + bv.w, 0.f) * (g4.w * sc) + b4.w;
  _Float16* xo = xout + (size_t)wid * DH + lane * 4;
  xo[0] = (_Float16)o0; xo[1] = (_Float16)o1;
  xo[2] = (_Float16)o2; xo[3] = (_Float16)o3;
}

// ---------------- WMMA GEMM: C[M,ncols] = A[M,128](f16) x Bt[n][k](f16), f32 acc ----
// one wave per 16x16 tile; K-loop of 4 x v_wmma_f32_16x16x32_f16
__global__ void k_wmma_gemm(const _Float16* __restrict__ A, const _Float16* __restrict__ Bt,
                            const float* __restrict__ bias, float* __restrict__ out32,
                            _Float16* __restrict__ out16, int ncols, int ldo, int relu) {
  int mtile = blockIdx.x;
  int ntile = blockIdx.y;
  int lane  = threadIdx.x & 31;
  int lm    = lane & 15;
  int half  = lane >> 4;

  const _Float16* arow = A  + (size_t)(mtile * 16 + lm) * DH;  // A: lane holds row m=lm
  const _Float16* brow = Bt + (size_t)(ntile * 16 + lm) * DH;  // B: lane holds col n=lm

  v8f c = {};
#pragma unroll
  for (int kb = 0; kb < DH; kb += 32) {
    // A 16x32 f16 layout: VGPR0-3 -> K = 8h+p (p=0..7); VGPR4-7 -> K = 16+8h+p
    v8h a0 = *(const v8h*)(arow + kb + 8 * half);
    v8h a1 = *(const v8h*)(arow + kb + 16 + 8 * half);
    // B 32x16 f16 layout: same K order as A: K = 8h+p (p=0..7), then K = 16+8h+p
    v8h b0 = *(const v8h*)(brow + kb + 8 * half);
    v8h b1 = *(const v8h*)(brow + kb + 16 + 8 * half);
    v16h av = __builtin_shufflevector(a0, a1, 0,1,2,3,4,5,6,7,8,9,10,11,12,13,14,15);
    v16h bv = __builtin_shufflevector(b0, b1, 0,1,2,3,4,5,6,7,8,9,10,11,12,13,14,15);
    c = __builtin_amdgcn_wmma_f32_16x16x32_f16(false, av, false, bv, (short)0, c,
                                               false, false);
  }

  int n = ntile * 16 + lm;
  float bval = 0.f;
  if (bias != nullptr && n < ncols) bval = bias[n];
#pragma unroll
  for (int r = 0; r < 8; ++r) {
    int m = mtile * 16 + r + 8 * half;   // C/D layout: VGPR r -> M = r (+8 for hi half)
    float v = c[r] + bval;
    if (relu) v = fmaxf(v, 0.f);
    if (out16) out16[(size_t)m * DH + n] = (_Float16)v;
    if (out32 && n < ncols) out32[(size_t)m * ldo + n] = v;
  }
}

// ---------------- host: orchestrate one forward pass ----------------
extern "C" void kernel_launch(void* const* d_in, const int* in_sizes, int n_in,
                              void* d_out, int out_size, void* d_ws, size_t ws_size,
                              hipStream_t stream) {
  (void)in_sizes; (void)n_in; (void)out_size; (void)ws_size;
  const float* pos = (const float*)d_in[0];
  const int*   ei  = (const int*)d_in[1];
  auto Wl  = [&](int l) { return (const float*)d_in[2 + (l - 1) * 6 + 0]; };
  auto aS  = [&](int l) { return (const float*)d_in[2 + (l - 1) * 6 + 1]; };
  auto aD  = [&](int l) { return (const float*)d_in[2 + (l - 1) * 6 + 2]; };
  auto bL  = [&](int l) { return (const float*)d_in[2 + (l - 1) * 6 + 3]; };
  auto bnG = [&](int l) { return (const float*)d_in[2 + (l - 1) * 6 + 4]; };
  auto bnB = [&](int l) { return (const float*)d_in[2 + (l - 1) * 6 + 5]; };
  const float* lin1_w = (const float*)d_in[26];
  const float* lin1_b = (const float*)d_in[27];
  const float* lin2_w = (const float*)d_in[28];
  const float* lin2_b = (const float*)d_in[29];
  const float* lin3_w = (const float*)d_in[30];
  const float* lin3_b = (const float*)d_in[31];
  float* out = (float*)d_out;

  // workspace carve-up (256B aligned slices)
  char* p = (char*)d_ws;
  auto carve = [&](size_t bytes) {
    void* r = (void*)p;
    p += (bytes + 255) & ~(size_t)255;
    return r;
  };
  _Float16* x16a = (_Float16*)carve((size_t)N_NODES * DH * 2);
  _Float16* x16b = (_Float16*)carve((size_t)N_NODES * DH * 2);
  float*    h32  = (float*)carve((size_t)N_NODES * DH * 4);
  float*    ews  = (float*)carve((size_t)EN * 4);
  int*      ssrc = (int*)carve((size_t)EN * 4);
  int*      offs = (int*)carve((size_t)(N_NODES + 1) * 4);
  int*      cnts = (int*)carve((size_t)N_NODES * 4);
  int*      curs = (int*)carve((size_t)N_NODES * 4);
  float*    als  = (float*)carve((size_t)N_NODES * 4);
  float*    ald  = (float*)carve((size_t)N_NODES * 4);
  _Float16* wt[6];
  for (int i = 0; i < 6; ++i) wt[i] = (_Float16*)carve((size_t)DH * DH * 2);

  const int TB = 256;
  // --- weight prep: transposed f16 copies (B-fragment friendly) ---
  {
    int total = DH * DH, g = (total + TB - 1) / TB;
    k_prep_weight<<<g, TB, 0, stream>>>(Wl(2), wt[0], DH, DH, DH);
    k_prep_weight<<<g, TB, 0, stream>>>(Wl(3), wt[1], DH, DH, DH);
    k_prep_weight<<<g, TB, 0, stream>>>(Wl(4), wt[2], DH, DH, DH);
    k_prep_weight<<<g, TB, 0, stream>>>(lin1_w, wt[3], DH, DH, DH);
    k_prep_weight<<<g, TB, 0, stream>>>(lin2_w, wt[4], DH, DH, DH);
    int total3 = DOUTP * DH, g3 = (total3 + TB - 1) / TB;
    k_prep_weight<<<g3, TB, 0, stream>>>(lin3_w, wt[5], DH, DOUT, DOUTP);
  }
  // --- CSR build (once; reused by all 4 layers) ---
  k_zero_counts<<<(N_NODES + TB - 1) / TB, TB, 0, stream>>>(cnts);
  k_histogram<<<(EN + TB - 1) / TB, TB, 0, stream>>>(ei, cnts);
  k_scan<<<1, 1024, 0, stream>>>(cnts, offs, curs);
  k_scatter<<<(EN + TB - 1) / TB, TB, 0, stream>>>(ei, curs, ssrc);

  const dim3 gemm_grid(N_NODES / 16, DH / 16);   // 3125 x 8
  const int  agg_blocks = (N_NODES * 32 + TB - 1) / TB;  // wave per node

  // --- layer 1 (K=3 scalar GEMM) ---
  k_l1_gemm<<<(N_NODES * DH + TB - 1) / TB, TB, 0, stream>>>(pos, Wl(1), h32);
  k_al<<<agg_blocks, TB, 0, stream>>>(h32, aS(1), aD(1), als, ald);
  k_aggregate<<<agg_blocks, TB, 0, stream>>>(offs, ssrc, ews, als, ald, h32,
                                             bL(1), bnG(1), bnB(1), x16a);
  // --- layers 2..4 (WMMA GEMM + attention + aggregation) ---
  for (int l = 2; l <= 4; ++l) {
    k_wmma_gemm<<<gemm_grid, 32, 0, stream>>>(x16a, wt[l - 2], nullptr, h32, nullptr,
                                              DH, DH, 0);
    k_al<<<agg_blocks, TB, 0, stream>>>(h32, aS(l), aD(l), als, ald);
    k_aggregate<<<agg_blocks, TB, 0, stream>>>(offs, ssrc, ews, als, ald, h32,
                                               bL(l), bnG(l), bnB(l), x16a);
  }
  // --- MLP head ---
  k_wmma_gemm<<<gemm_grid, 32, 0, stream>>>(x16a, wt[3], lin1_b, nullptr, x16b,
                                            DH, DH, 1);                 // relu
  k_wmma_gemm<<<gemm_grid, 32, 0, stream>>>(x16b, wt[4], lin2_b, nullptr, x16a,
                                            DH, DH, 0);
  const dim3 gemm3_grid(N_NODES / 16, DOUTP / 16);                      // 3125 x 7
  k_wmma_gemm<<<gemm3_grid, 32, 0, stream>>>(x16a, wt[5], lin3_b, out, nullptr,
                                             DOUT, DOUT, 0);
}
